// HexagramMoE_44384192037445
// MI455X (gfx1250) — compile-verified, matrix operand order checked by reading, not run on hardware
//
#include <hip/hip_runtime.h>
#include <hip/hip_bf16.h>

#define NTOK 8192
#define DIM  1024
#define HDIM 512

typedef __attribute__((ext_vector_type(16))) __bf16 v16bf;
typedef __attribute__((ext_vector_type(8)))  float  v8f;
typedef __attribute__((ext_vector_type(8)))  unsigned short u16x8;

union Frag16 {
    v16bf bf;
    u16x8 us[2];
};

__device__ __forceinline__ unsigned short f2bf(float f) {
    unsigned int u = __float_as_uint(f);
    u += 0x7FFFu + ((u >> 16) & 1u);      // round-to-nearest-even
    return (unsigned short)(u >> 16);
}

// ---------------------------------------------------------------- init
__global__ void hexmoe_init_kernel(float* probsum) {
    if (threadIdx.x < 64) probsum[threadIdx.x] = 0.0f;
}

// ---------------------------------------------------------------- router
// One wave32 per token; 8 waves / block.
__global__ void hexmoe_router_kernel(const float* __restrict__ x,
                                     const float* __restrict__ rw,
                                     const float* __restrict__ log_temp,
                                     float* __restrict__ tokw,
                                     float* __restrict__ probsum) {
    __shared__ float lds_prob[64];
    const int lane = threadIdx.x & 31;
    const int wave = threadIdx.x >> 5;
    if (threadIdx.x < 64) lds_prob[threadIdx.x] = 0.0f;
    __syncthreads();

    const int token = blockIdx.x * 8 + wave;
    const float* xrow = x + (size_t)token * DIM;

    // z = x @ router_w.T  (per-lane partials, strided by 32 -> coalesced)
    float z[6];
#pragma unroll
    for (int j = 0; j < 6; j++) z[j] = 0.0f;
    for (int d = lane; d < DIM; d += 32) {
        float xv = xrow[d];
#pragma unroll
        for (int j = 0; j < 6; j++) z[j] += xv * rw[j * DIM + d];
    }
#pragma unroll
    for (int j = 0; j < 6; j++)
        for (int off = 16; off >= 1; off >>= 1)
            z[j] += __shfl_xor(z[j], off, 32);

    float temp = __expf(*log_temp);
    temp = fminf(fmaxf(temp, 0.01f), 5.0f);

    // Each lane owns hexagram codes {lane, lane+32}
    float lg[2];
#pragma unroll
    for (int t = 0; t < 2; t++) {
        int c = lane + t * 32;
        float d2 = 0.0f;
#pragma unroll
        for (int j = 0; j < 6; j++) {
            float bit = (float)((c >> (5 - j)) & 1);
            float df = z[j] - bit;
            d2 += df * df;
        }
        lg[t] = -sqrtf(d2) / temp;
    }

    // iterative wave-wide top-4 (tie-break: lowest index, like jax top_k)
    float topv[4];
    int   topi[4];
    bool used0 = false, used1 = false;
#pragma unroll
    for (int k = 0; k < 4; k++) {
        float bv = -1e30f; int bi = 0x7fffffff;
        if (!used0 && lg[0] > bv) { bv = lg[0]; bi = lane; }
        if (!used1 && (lg[1] > bv || (lg[1] == bv && lane + 32 < bi))) { bv = lg[1]; bi = lane + 32; }
        for (int off = 16; off >= 1; off >>= 1) {
            float ov = __shfl_xor(bv, off, 32);
            int   oi = __shfl_xor(bi, off, 32);
            if (ov > bv || (ov == bv && oi < bi)) { bv = ov; bi = oi; }
        }
        topv[k] = bv; topi[k] = bi;
        if (bi == lane)      used0 = true;
        if (bi == lane + 32) used1 = true;
    }

    // softmax over top-4 (topv[0] is the max)
    float ek[4], se = 0.0f;
#pragma unroll
    for (int k = 0; k < 4; k++) { ek[k] = __expf(topv[k] - topv[0]); se += ek[k]; }
    float inv_se = 1.0f / se;

    // fold top-4 into per-(token, expert) weights: [0..7]=upper, [8..15]=lower
    if (lane < 16) {
        float wv = 0.0f;
#pragma unroll
        for (int k = 0; k < 4; k++) {
            float wk = ek[k] * inv_se;
            int ue = topi[k] >> 3, le = topi[k] & 7;
            if (lane < 8) { if (ue == lane)     wv += wk; }
            else          { if (le == lane - 8) wv += wk; }
        }
        tokw[(size_t)token * 16 + lane] = wv;
    }

    // balance: full softmax over 64 codes, accumulate mean probs
    float p0 = __expf(lg[0] - topv[0]);
    float p1 = __expf(lg[1] - topv[0]);
    float ps = p0 + p1;
    for (int off = 16; off >= 1; off >>= 1) ps += __shfl_xor(ps, off, 32);
    float inv = 1.0f / ps;
    atomicAdd(&lds_prob[lane],      p0 * inv);
    atomicAdd(&lds_prob[lane + 32], p1 * inv);
    __syncthreads();
    if (threadIdx.x < 64) atomicAdd(&probsum[threadIdx.x], lds_prob[threadIdx.x]);
}

// ---------------------------------------------------------------- balance
__global__ void hexmoe_balance_kernel(const float* __restrict__ probsum,
                                      float* __restrict__ out_balance) {
    __shared__ float s[64];
    int t = threadIdx.x;
    s[t] = logf(probsum[t] / (float)NTOK);
    __syncthreads();
    if (t == 0) {
        float sum = 0.0f;
        for (int i = 0; i < 64; i++) sum += s[i];
        float lu = logf(1.0f / 64.0f);
        out_balance[0] = 0.01f * (lu - sum * (1.0f / 64.0f));
    }
}

// ---------------------------------------------------------------- converts
__global__ void hexmoe_cvt_x_kernel(const float* __restrict__ src,
                                    unsigned short* __restrict__ dst, long total) {
    for (long i = (long)blockIdx.x * blockDim.x + threadIdx.x; i < total;
         i += (long)gridDim.x * blockDim.x)
        dst[i] = f2bf(src[i]);
}

// w1t[e, h, d] = w1[e%8, d, h]   (dst: [16, HDIM, DIM])
__global__ void hexmoe_cvt_w1_kernel(const float* __restrict__ uw1,
                                     const float* __restrict__ lw1,
                                     unsigned short* __restrict__ dst) {
    const long total = 16L * HDIM * DIM;
    for (long i = (long)blockIdx.x * blockDim.x + threadIdx.x; i < total;
         i += (long)gridDim.x * blockDim.x) {
        int e = (int)(i / (HDIM * DIM));
        int rem = (int)(i % (HDIM * DIM));
        int h = rem / DIM, d = rem % DIM;
        const float* src = (e < 8) ? uw1 : lw1;
        dst[i] = f2bf(src[((long)(e & 7) * DIM + d) * HDIM + h]);
    }
}

// w2t[e, d, h] = w2[e%8, h, d]   (dst: [16, DIM, HDIM])
__global__ void hexmoe_cvt_w2_kernel(const float* __restrict__ uw2,
                                     const float* __restrict__ lw2,
                                     unsigned short* __restrict__ dst) {
    const long total = 16L * DIM * HDIM;
    for (long i = (long)blockIdx.x * blockDim.x + threadIdx.x; i < total;
         i += (long)gridDim.x * blockDim.x) {
        int e = (int)(i / (DIM * HDIM));
        int rem = (int)(i % (DIM * HDIM));
        int d = rem / HDIM, h = rem % HDIM;
        const float* src = (e < 8) ? uw2 : lw2;
        dst[i] = f2bf(src[((long)(e & 7) * HDIM + h) * DIM + d]);
    }
}

// ---------------------------------------------------------------- GEMM1: H = silu(X @ W1_e)
// grid (HDIM/64, NTOK/128), block 128 (4 waves).
// Wave w: 32 tokens [mbase, mbase+32) x 64 h-cols -> 8 accumulators; each B frag feeds 2 wmmas.
__global__ void hexmoe_gemm1_kernel(const unsigned short* __restrict__ xb,   // [NTOK, DIM]
                                    const unsigned short* __restrict__ w1t,  // [16, HDIM, DIM]
                                    unsigned short* __restrict__ hbuf,       // [NTOK, HDIM]
                                    int expert) {
    const int lane = threadIdx.x & 31;
    const int wave = threadIdx.x >> 5;
    const int hbase = blockIdx.x * 64;
    const int mbase = blockIdx.y * 128 + wave * 32;
    const int l15 = lane & 15;
    const int khalf = (lane >> 4) * 8;   // ISA 16-bit A/B fragment K split

    const unsigned short* xrow0 = xb + (size_t)(mbase + l15) * DIM;
    const unsigned short* xrow1 = xrow0 + (size_t)16 * DIM;
    const unsigned short* wbase = w1t + (size_t)expert * HDIM * DIM;

    v8f acc[2][4];
#pragma unroll
    for (int s = 0; s < 2; s++)
#pragma unroll
        for (int i = 0; i < 4; i++) acc[s][i] = {};

    for (int k = 0; k < DIM; k += 32) {
        __builtin_prefetch(xrow0 + k + 128, 0, 0);   // global_prefetch next A slice
        Frag16 a0, a1;
        a0.us[0] = *(const u16x8*)(xrow0 + k + khalf);
        a0.us[1] = *(const u16x8*)(xrow0 + k + khalf + 16);
        a1.us[0] = *(const u16x8*)(xrow1 + k + khalf);
        a1.us[1] = *(const u16x8*)(xrow1 + k + khalf + 16);
#pragma unroll
        for (int nt = 0; nt < 4; nt++) {
            const unsigned short* wrow = wbase + (size_t)(hbase + nt * 16 + l15) * DIM;
            Frag16 b;
            b.us[0] = *(const u16x8*)(wrow + k + khalf);
            b.us[1] = *(const u16x8*)(wrow + k + khalf + 16);
            acc[0][nt] = __builtin_amdgcn_wmma_f32_16x16x32_bf16(
                false, a0.bf, false, b.bf, (short)0, acc[0][nt], false, false);
            acc[1][nt] = __builtin_amdgcn_wmma_f32_16x16x32_bf16(
                false, a1.bf, false, b.bf, (short)0, acc[1][nt], false, false);
        }
    }

    // C layout: lane -> col = l15, rows = r + (lane<16 ? 0 : 8)
#pragma unroll
    for (int s = 0; s < 2; s++) {
        const int mrow0 = mbase + s * 16 + ((lane < 16) ? 0 : 8);
#pragma unroll
        for (int nt = 0; nt < 4; nt++) {
            int col = hbase + nt * 16 + l15;
#pragma unroll
            for (int r = 0; r < 8; r++) {
                float v = acc[s][nt][r];
                float sg = v / (1.0f + __expf(-v));     // silu
                hbuf[(size_t)(mrow0 + r) * HDIM + col] = f2bf(sg);
            }
        }
    }
}

// ---------------------------------------------------------------- GEMM2: out += tokw[:,e] * (H @ W2_e)
// grid (DIM/64, NTOK/128), block 128 (4 waves); same 32x64 per-wave tiling.
__global__ void hexmoe_gemm2_kernel(const unsigned short* __restrict__ hbuf,  // [NTOK, HDIM]
                                    const unsigned short* __restrict__ w2t,   // [16, DIM, HDIM]
                                    const float* __restrict__ tokw,           // [NTOK, 16]
                                    float* __restrict__ out,                  // [NTOK, DIM]
                                    int expert, int accumulate) {
    const int lane = threadIdx.x & 31;
    const int wave = threadIdx.x >> 5;
    const int dbase = blockIdx.x * 64;
    const int mbase = blockIdx.y * 128 + wave * 32;
    const int l15 = lane & 15;
    const int khalf = (lane >> 4) * 8;

    const unsigned short* hrow0 = hbuf + (size_t)(mbase + l15) * HDIM;
    const unsigned short* hrow1 = hrow0 + (size_t)16 * HDIM;
    const unsigned short* wbase = w2t + (size_t)expert * DIM * HDIM;

    v8f acc[2][4];
#pragma unroll
    for (int s = 0; s < 2; s++)
#pragma unroll
        for (int i = 0; i < 4; i++) acc[s][i] = {};

    for (int k = 0; k < HDIM; k += 32) {
        __builtin_prefetch(hrow0 + k + 128, 0, 0);
        Frag16 a0, a1;
        a0.us[0] = *(const u16x8*)(hrow0 + k + khalf);
        a0.us[1] = *(const u16x8*)(hrow0 + k + khalf + 16);
        a1.us[0] = *(const u16x8*)(hrow1 + k + khalf);
        a1.us[1] = *(const u16x8*)(hrow1 + k + khalf + 16);
#pragma unroll
        for (int nt = 0; nt < 4; nt++) {
            const unsigned short* wrow = wbase + (size_t)(dbase + nt * 16 + l15) * HDIM;
            Frag16 b;
            b.us[0] = *(const u16x8*)(wrow + k + khalf);
            b.us[1] = *(const u16x8*)(wrow + k + khalf + 16);
            acc[0][nt] = __builtin_amdgcn_wmma_f32_16x16x32_bf16(
                false, a0.bf, false, b.bf, (short)0, acc[0][nt], false, false);
            acc[1][nt] = __builtin_amdgcn_wmma_f32_16x16x32_bf16(
                false, a1.bf, false, b.bf, (short)0, acc[1][nt], false, false);
        }
    }

#pragma unroll
    for (int s = 0; s < 2; s++) {
        const int mrow0 = mbase + s * 16 + ((lane < 16) ? 0 : 8);
        float wgt[8];
#pragma unroll
        for (int r = 0; r < 8; r++) wgt[r] = tokw[(size_t)(mrow0 + r) * 16 + expert];
#pragma unroll
        for (int nt = 0; nt < 4; nt++) {
            int col = dbase + nt * 16 + l15;
#pragma unroll
            for (int r = 0; r < 8; r++) {
                size_t idx = (size_t)(mrow0 + r) * DIM + col;
                float v = wgt[r] * acc[s][nt][r];
                if (accumulate) v += out[idx];
                out[idx] = v;
            }
        }
    }
}

// ---------------------------------------------------------------- launch
extern "C" void kernel_launch(void* const* d_in, const int* in_sizes, int n_in,
                              void* d_out, int out_size, void* d_ws, size_t ws_size,
                              hipStream_t stream) {
    (void)in_sizes; (void)n_in; (void)out_size; (void)ws_size;

    const float* x        = (const float*)d_in[0];
    const float* router_w = (const float*)d_in[1];
    const float* log_temp = (const float*)d_in[2];
    const float* uw1      = (const float*)d_in[3];
    const float* uw2      = (const float*)d_in[4];
    const float* lw1      = (const float*)d_in[5];
    const float* lw2      = (const float*)d_in[6];
    float* out = (float*)d_out;

    // workspace layout (bytes)
    char* ws = (char*)d_ws;
    unsigned short* xb   = (unsigned short*)(ws);                       // 16 MB
    unsigned short* w1t  = (unsigned short*)(ws + (16L << 20));         // 16 MB
    unsigned short* w2t  = (unsigned short*)(ws + (32L << 20));         // 16 MB
    unsigned short* hbuf = (unsigned short*)(ws + (48L << 20));         // 8 MB
    float*          tokw = (float*)         (ws + (56L << 20));         // 512 KB
    float*          psum = (float*)         (ws + (57L << 20));         // 256 B

    hipLaunchKernelGGL(hexmoe_init_kernel, dim3(1), dim3(64), 0, stream, psum);
    hipLaunchKernelGGL(hexmoe_router_kernel, dim3(NTOK / 8), dim3(256), 0, stream,
                       x, router_w, log_temp, tokw, psum);

    hipLaunchKernelGGL(hexmoe_cvt_x_kernel, dim3(2048), dim3(256), 0, stream,
                       x, xb, (long)NTOK * DIM);
    hipLaunchKernelGGL(hexmoe_cvt_w1_kernel, dim3(2048), dim3(256), 0, stream, uw1, lw1, w1t);
    hipLaunchKernelGGL(hexmoe_cvt_w2_kernel, dim3(2048), dim3(256), 0, stream, uw2, lw2, w2t);

    for (int e = 0; e < 16; e++) {
        hipLaunchKernelGGL(hexmoe_gemm1_kernel, dim3(HDIM / 64, NTOK / 128), dim3(128), 0, stream,
                           xb, w1t, hbuf, e);
        hipLaunchKernelGGL(hexmoe_gemm2_kernel, dim3(DIM / 64, NTOK / 128), dim3(128), 0, stream,
                           hbuf, w2t, tokw, out, e, (e > 0) ? 1 : 0);
    }

    hipLaunchKernelGGL(hexmoe_balance_kernel, dim3(1), dim3(64), 0, stream,
                       psum, out + (size_t)NTOK * DIM);
}
